// MultiHeadAttention_12206297055296
// MI455X (gfx1250) — matrix operand order repaired and checked by reading, not compiled, not run on hardware
//
#include <hip/hip_runtime.h>
#include <hip/hip_bf16.h>

// Problem dims (fixed by reference)
#define BB   2
#define TT   2048
#define DD   1024
#define HH   16
#define HDIM 64
#define BT   (BB * TT)   // 4096

typedef __attribute__((ext_vector_type(16))) __bf16 v16bf;
typedef __attribute__((ext_vector_type(8)))  float  v8f;

union ABTile { unsigned int u[8]; v16bf v; };

__device__ __forceinline__ unsigned short f32_to_bf16_bits(float f) {
    unsigned int x = __float_as_uint(f);
    unsigned int r = x + 0x7FFFu + ((x >> 16) & 1u);  // round-to-nearest-even
    return (unsigned short)(r >> 16);
}

// A operand (16x32, M x K, bf16): VGPR j<4 holds K = half*8 + 2j(+1),
// VGPR j>=4 holds K = 16 + half*8 + 2(j-4)(+1).   [ISA 7.12.2]
__device__ __forceinline__ int a_kidx(int j, int half) {
    return (j < 4) ? (half * 8 + 2 * j) : (16 + half * 8 + 2 * (j - 4));
}
// B operand (32x16, K x N, bf16): same K order as A: VGPR j<4 holds K = half*8 + 2j(+1),
// VGPR j>=4 holds K = 16 + half*8 + 2(j-4)(+1).
__device__ __forceinline__ int b_kidx(int j, int half) {
    return (j < 4) ? (half * 8 + 2 * j) : (16 + half * 8 + 2 * (j - 4));
}

// ---------------- pack / transpose kernels ----------------

__global__ void k_pack(const float* __restrict__ src,
                       unsigned short* __restrict__ dst, int n) {
    int i = blockIdx.x * blockDim.x + threadIdx.x;
    if (i < n) dst[i] = f32_to_bf16_bits(src[i]);
}

// W [H, D, HD] f32 -> Wt [H, HD, D] bf16 (K contiguous for B operand)
__global__ void k_pack_w_t(const float* __restrict__ src,
                           unsigned short* __restrict__ dst, int n) {
    int i = blockIdx.x * blockDim.x + threadIdx.x;
    if (i >= n) return;
    int hd = i % HDIM;
    int d  = (i / HDIM) % DD;
    int h  = i / (DD * HDIM);
    dst[((size_t)h * HDIM + hd) * DD + d] = f32_to_bf16_bits(src[i]);
}

// Wo [D_in, D_out] f32 -> WoT [D_out, D_in] bf16
__global__ void k_pack_wo_t(const float* __restrict__ src,
                            unsigned short* __restrict__ dst, int n) {
    int i = blockIdx.x * blockDim.x + threadIdx.x;
    if (i >= n) return;
    int o = i % DD;
    int d = i / DD;
    dst[(size_t)o * DD + d] = f32_to_bf16_bits(src[i]);
}

// ---------------- QKV projection: X[BT,D] @ W[h][D,HD] ----------------
// grid (BT/16, H, 3), block 32. z: 0=Q,1=K,2=V. V stored transposed [B,H,HD,T].

__global__ void __launch_bounds__(32) k_qkv(
    const unsigned short* __restrict__ xb,   // [BT, D] bf16
    const unsigned short* __restrict__ wqt,  // [H, HD, D] bf16
    const unsigned short* __restrict__ wkt,
    const unsigned short* __restrict__ wvt,
    unsigned short* __restrict__ qout,       // [B,H,T,HD]
    unsigned short* __restrict__ kout,       // [B,H,T,HD]
    unsigned short* __restrict__ vtout)      // [B,H,HD,T]
{
    const int lane = threadIdx.x;
    const int half = lane >> 4;
    const int ln   = lane & 15;
    const int tile = blockIdx.x;
    const int h    = blockIdx.y;
    const int z    = blockIdx.z;
    const unsigned short* w = (z == 0) ? wqt : (z == 1) ? wkt : wvt;
    const int row0 = tile * 16;

    v8f c[4];
    #pragma unroll
    for (int i = 0; i < 4; ++i)
        #pragma unroll
        for (int e = 0; e < 8; ++e) c[i][e] = 0.f;

    for (int kk = 0; kk < DD; kk += 32) {
        ABTile a;
        const unsigned short* arow = xb + (size_t)(row0 + ln) * DD + kk;
        #pragma unroll
        for (int j = 0; j < 8; ++j)
            a.u[j] = *(const unsigned int*)(arow + a_kidx(j, half));
        #pragma unroll
        for (int nt = 0; nt < 4; ++nt) {
            ABTile bt;
            const unsigned short* brow =
                w + ((size_t)h * HDIM + nt * 16 + ln) * DD + kk;
            #pragma unroll
            for (int j = 0; j < 8; ++j)
                bt.u[j] = *(const unsigned int*)(brow + b_kidx(j, half));
            c[nt] = __builtin_amdgcn_wmma_f32_16x16x32_bf16(
                false, a.v, false, bt.v, (short)0, c[nt], false, false);
        }
    }

    const int b  = row0 / TT;
    const int t0 = row0 % TT;
    const size_t bh = (size_t)(b * HH + h);
    #pragma unroll
    for (int nt = 0; nt < 4; ++nt) {
        #pragma unroll
        for (int r = 0; r < 8; ++r) {
            int m  = r + 8 * half;
            int hd = nt * 16 + ln;
            unsigned short val = f32_to_bf16_bits(c[nt][r]);
            if (z == 0)
                qout[(bh * TT + t0 + m) * HDIM + hd] = val;
            else if (z == 1)
                kout[(bh * TT + t0 + m) * HDIM + hd] = val;
            else
                vtout[(bh * HDIM + hd) * TT + t0 + m] = val;
        }
    }
}

// ---------------- Flash attention (causal, online softmax) ----------------
// grid (T/16, H, B), block 32. One wave owns a 16-query tile; streams keys
// in blocks of 32 with WMMA Q@K^T and P@V, online softmax in registers.

__global__ void __launch_bounds__(32) k_attn(
    const unsigned short* __restrict__ Q,   // [B,H,T,HD]
    const unsigned short* __restrict__ K,   // [B,H,T,HD]
    const unsigned short* __restrict__ Vt,  // [B,H,HD,T]
    unsigned short* __restrict__ AO)        // [B,T,H*HD]
{
    __shared__ unsigned short p_lds[16][32];  // P tile bounce: C-layout -> A-layout

    const int lane = threadIdx.x;
    const int half = lane >> 4;
    const int ln   = lane & 15;
    const int qt   = blockIdx.x;
    const int h    = blockIdx.y;
    const int b    = blockIdx.z;
    const size_t bh = (size_t)(b * HH + h);
    const unsigned short* Qh = Q  + bh * TT * HDIM;
    const unsigned short* Kh = K  + bh * TT * HDIM;
    const unsigned short* Vh = Vt + bh * HDIM * TT;

    // Q A-operands for head-dim halves K=0..31 and K=32..63 (loaded once)
    ABTile qa[2];
    {
        const unsigned short* qrow = Qh + (size_t)(qt * 16 + ln) * HDIM;
        #pragma unroll
        for (int kh = 0; kh < 2; ++kh)
            #pragma unroll
            for (int j = 0; j < 8; ++j)
                qa[kh].u[j] =
                    *(const unsigned int*)(qrow + kh * 32 + a_kidx(j, half));
    }

    float m_run[8], l_run[8];
    v8f acc[4];
    #pragma unroll
    for (int r = 0; r < 8; ++r) { m_run[r] = -1e30f; l_run[r] = 0.f; }
    #pragma unroll
    for (int i = 0; i < 4; ++i)
        #pragma unroll
        for (int e = 0; e < 8; ++e) acc[i][e] = 0.f;

    const int q_end = qt * 16 + 15;
    for (int s0 = 0; s0 <= q_end; s0 += 32) {
        // ---- scores: two 16x16 C tiles over a 32-key block ----
        v8f s[2];
        #pragma unroll
        for (int st = 0; st < 2; ++st) {
            #pragma unroll
            for (int e = 0; e < 8; ++e) s[st][e] = 0.f;
            #pragma unroll
            for (int kh = 0; kh < 2; ++kh) {
                ABTile bt;
                const unsigned short* krow =
                    Kh + (size_t)(s0 + st * 16 + ln) * HDIM + kh * 32;
                #pragma unroll
                for (int j = 0; j < 8; ++j)
                    bt.u[j] = *(const unsigned int*)(krow + b_kidx(j, half));
                s[st] = __builtin_amdgcn_wmma_f32_16x16x32_bf16(
                    false, qa[kh].v, false, bt.v, (short)0, s[st], false, false);
            }
        }

        // ---- scale, causal mask, online softmax ----
        #pragma unroll
        for (int r = 0; r < 8; ++r) {
            const int qpos = qt * 16 + r + 8 * half;   // C layout: row = r + 8*half
            float mx = -1e30f;
            #pragma unroll
            for (int st = 0; st < 2; ++st) {
                const int kpos = s0 + st * 16 + ln;    // C layout: col = lane&15
                float val = s[st][r] * 0.125f;         // 1/sqrt(64)
                if (kpos > qpos) val = -1e30f;
                s[st][r] = val;
                mx = fmaxf(mx, val);
            }
            // row reduction across the 16 lanes of this half
            #pragma unroll
            for (int off = 1; off < 16; off <<= 1)
                mx = fmaxf(mx, __shfl_xor(mx, off, 32));
            const float m_new = fmaxf(m_run[r], mx);
            const float scale = __expf(m_run[r] - m_new);
            m_run[r] = m_new;

            float rs = 0.f;
            #pragma unroll
            for (int st = 0; st < 2; ++st) {
                float p = __expf(s[st][r] - m_new);
                s[st][r] = p;
                rs += p;
            }
            #pragma unroll
            for (int off = 1; off < 16; off <<= 1)
                rs += __shfl_xor(rs, off, 32);
            l_run[r] = l_run[r] * scale + rs;
            #pragma unroll
            for (int nt = 0; nt < 4; ++nt) acc[nt][r] *= scale;
        }

        // ---- P: C-layout f32 -> A-layout bf16 via LDS bounce ----
        #pragma unroll
        for (int st = 0; st < 2; ++st)
            #pragma unroll
            for (int r = 0; r < 8; ++r)
                p_lds[r + 8 * half][st * 16 + ln] = f32_to_bf16_bits(s[st][r]);
        __syncthreads();
        ABTile pa;
        #pragma unroll
        for (int j = 0; j < 8; ++j)
            pa.u[j] = *(const unsigned int*)(&p_lds[ln][a_kidx(j, half)]);
        __syncthreads();

        // ---- O += P(16x32) @ V(32x64): V^T layout gives K-contiguous B ----
        #pragma unroll
        for (int nt = 0; nt < 4; ++nt) {
            ABTile bt;
            const unsigned short* vrow =
                Vh + (size_t)(nt * 16 + ln) * TT + s0;
            #pragma unroll
            for (int j = 0; j < 8; ++j)
                bt.u[j] = *(const unsigned int*)(vrow + b_kidx(j, half));
            acc[nt] = __builtin_amdgcn_wmma_f32_16x16x32_bf16(
                false, pa.v, false, bt.v, (short)0, acc[nt], false, false);
        }
    }

    // ---- normalize and write AO [B,T,H*HD] bf16 ----
    #pragma unroll
    for (int nt = 0; nt < 4; ++nt) {
        #pragma unroll
        for (int r = 0; r < 8; ++r) {
            int m = r + 8 * half;
            float o = acc[nt][r] / l_run[r];
            AO[((size_t)(b * TT + qt * 16 + m)) * DD + h * HDIM + nt * 16 + ln] =
                f32_to_bf16_bits(o);
        }
    }
}

// ---------------- Output projection: AO[BT,D] @ Wo[D,D] + bo ----------------
// grid (BT/16, D/64), block 32.

__global__ void __launch_bounds__(32) k_proj(
    const unsigned short* __restrict__ AO,   // [BT, D] bf16
    const unsigned short* __restrict__ WoT,  // [D_out, D_in] bf16
    const float* __restrict__ bo,            // [D]
    float* __restrict__ out)                 // [BT, D] f32
{
    const int lane = threadIdx.x;
    const int half = lane >> 4;
    const int ln   = lane & 15;
    const int tile = blockIdx.x;
    const int ot   = blockIdx.y;
    const int row0 = tile * 16;

    v8f c[4];
    #pragma unroll
    for (int i = 0; i < 4; ++i)
        #pragma unroll
        for (int e = 0; e < 8; ++e) c[i][e] = 0.f;

    for (int kk = 0; kk < DD; kk += 32) {
        ABTile a;
        const unsigned short* arow = AO + (size_t)(row0 + ln) * DD + kk;
        #pragma unroll
        for (int j = 0; j < 8; ++j)
            a.u[j] = *(const unsigned int*)(arow + a_kidx(j, half));
        #pragma unroll
        for (int nt = 0; nt < 4; ++nt) {
            ABTile bt;
            const unsigned short* brow =
                WoT + (size_t)(ot * 64 + nt * 16 + ln) * DD + kk;
            #pragma unroll
            for (int j = 0; j < 8; ++j)
                bt.u[j] = *(const unsigned int*)(brow + b_kidx(j, half));
            c[nt] = __builtin_amdgcn_wmma_f32_16x16x32_bf16(
                false, a.v, false, bt.v, (short)0, c[nt], false, false);
        }
    }

    #pragma unroll
    for (int nt = 0; nt < 4; ++nt) {
        #pragma unroll
        for (int r = 0; r < 8; ++r) {
            int m   = r + 8 * half;
            int col = ot * 64 + nt * 16 + ln;
            out[(size_t)(row0 + m) * DD + col] = c[nt][r] + bo[col];
        }
    }
}

// ---------------- host launcher ----------------

extern "C" void kernel_launch(void* const* d_in, const int* in_sizes, int n_in,
                              void* d_out, int out_size, void* d_ws, size_t ws_size,
                              hipStream_t stream) {
    const float* x  = (const float*)d_in[0];
    const float* Wq = (const float*)d_in[1];
    const float* Wk = (const float*)d_in[2];
    const float* Wv = (const float*)d_in[3];
    const float* Wo = (const float*)d_in[4];
    const float* bo = (const float*)d_in[5];
    float* out = (float*)d_out;

    unsigned short* ws = (unsigned short*)d_ws;
    size_t off = 0;
    auto alloc = [&](size_t elems) { unsigned short* p = ws + off; off += elems; return p; };

    unsigned short* xb  = alloc((size_t)BT * DD);             // x bf16
    unsigned short* wqt = alloc((size_t)HH * HDIM * DD);      // Wq^T
    unsigned short* wkt = alloc((size_t)HH * HDIM * DD);      // Wk^T
    unsigned short* wvt = alloc((size_t)HH * HDIM * DD);      // Wv^T
    unsigned short* wot = alloc((size_t)DD * DD);             // Wo^T
    unsigned short* Qb  = alloc((size_t)BB * HH * TT * HDIM); // Q
    unsigned short* Kb  = alloc((size_t)BB * HH * TT * HDIM); // K
    unsigned short* Vtb = alloc((size_t)BB * HH * TT * HDIM); // V transposed
    unsigned short* AOb = alloc((size_t)BT * DD);             // attn out

    const int nx = BT * DD;
    k_pack<<<(nx + 255) / 256, 256, 0, stream>>>(x, xb, nx);
    const int nw = HH * DD * HDIM;
    k_pack_w_t<<<(nw + 255) / 256, 256, 0, stream>>>(Wq, wqt, nw);
    k_pack_w_t<<<(nw + 255) / 256, 256, 0, stream>>>(Wk, wkt, nw);
    k_pack_w_t<<<(nw + 255) / 256, 256, 0, stream>>>(Wv, wvt, nw);
    const int no = DD * DD;
    k_pack_wo_t<<<(no + 255) / 256, 256, 0, stream>>>(Wo, wot, no);

    k_qkv<<<dim3(BT / 16, HH, 3), 32, 0, stream>>>(xb, wqt, wkt, wvt, Qb, Kb, Vtb);
    k_attn<<<dim3(TT / 16, HH, BB), 32, 0, stream>>>(Qb, Kb, Vtb, AOb);
    k_proj<<<dim3(BT / 16, DD / 64), 32, 0, stream>>>(AOb, wot, bo, out);
}
